// MegatronAttention_5841155523044
// MI455X (gfx1250) — matrix operand order repaired and checked by reading, not compiled, not run on hardware
//
#include <hip/hip_runtime.h>
#include <hip/hip_bf16.h>
#include <math.h>

// ---------------------------------------------------------------------------
// MI455X / gfx1250: wave32 + v_wmma_f32_16x16x32_bf16 pipeline for MHA block.
// ---------------------------------------------------------------------------

typedef __bf16 bhalf;
typedef __bf16 bhalf8  __attribute__((ext_vector_type(8)));
typedef __bf16 bhalf16 __attribute__((ext_vector_type(16)));
typedef float  floatx8 __attribute__((ext_vector_type(8)));

#define S_LEN   2048
#define H_DIM   2048
#define NHEADS  32
#define NGRP    8
#define HD      64
#define QKV_N   3072
#define QSZ     2048   // Q columns in qkv
#define KVSZ    512    // K (or V) columns in qkv

// A-matrix 16x32 bf16 fragment (ISA 7.12.2):
//   lanes 0-15 : M=lane,  K = 0..7 (v0-3) and 16..23 (v4-7)
//   lanes16-31 : M=lane-16, K = 8..15 and 24..31
__device__ __forceinline__ bhalf16 load_a16x32(const bhalf* base, int ld, int lane) {
    const int m = lane & 15, h = (lane >> 4) & 1;
    bhalf8 lo = *reinterpret_cast<const bhalf8*>(base + (size_t)m * ld + h * 8);
    bhalf8 hi = *reinterpret_cast<const bhalf8*>(base + (size_t)m * ld + 16 + h * 8);
    return __builtin_shufflevector(lo, hi, 0,1,2,3,4,5,6,7,8,9,10,11,12,13,14,15);
}

// B-matrix 32x16 bf16 fragment: lane n holds B[k][n] in the A-fragment K order
// (lanes 0-15: K=0..7,16..23, lanes 16-31: K=8..15,24..31).  With B = W^T for row-major
// W[n][k], each lane reads two runs of 8 contiguous bf16 from row n of W.
__device__ __forceinline__ bhalf16 load_b32x16(const bhalf* base, int ld, int lane) {
    const int n = lane & 15, h = (lane >> 4) & 1;
    bhalf8 lo = *reinterpret_cast<const bhalf8*>(base + (size_t)n * ld + h * 8);
    bhalf8 hi = *reinterpret_cast<const bhalf8*>(base + (size_t)n * ld + 16 + h * 8);
    return __builtin_shufflevector(lo, hi, 0,1,2,3,4,5,6,7,8,9,10,11,12,13,14,15);
}

__device__ __forceinline__ floatx8 wmma_bf16(bhalf16 a, bhalf16 b, floatx8 c) {
    return __builtin_amdgcn_wmma_f32_16x16x32_bf16(
        /*neg_a=*/false, a, /*neg_b=*/false, b,
        /*c_mod=*/(short)0, c, /*reuse_a=*/false, /*reuse_b=*/false);
}

// ---------------------------------------------------------------------------
// Kernel 1: fp32 -> bf16 cast (grid-stride)
// ---------------------------------------------------------------------------
__global__ void cast_f32_bf16(const float* __restrict__ src, bhalf* __restrict__ dst, int n) {
    int i = blockIdx.x * blockDim.x + threadIdx.x;
    int stride = gridDim.x * blockDim.x;
    for (; i < n; i += stride) dst[i] = (bhalf)src[i];
}

// ---------------------------------------------------------------------------
// Kernel 2: qkv = X @ Wqkv^T + bqkv   (X:[S,H] bf16, Wqkv:[3072,H] bf16)
// Each wave computes a 32x32 output tile (2x2 WMMA accumulators).
// Also scatters V columns transposed into vT[NGRP*HD, S] for the PV GEMM.
// ---------------------------------------------------------------------------
__global__ void gemm_qkv_kernel(const bhalf* __restrict__ X, const bhalf* __restrict__ W,
                                const float* __restrict__ bias,
                                bhalf* __restrict__ qkv, bhalf* __restrict__ vT) {
    const int lane = threadIdx.x & 31;
    const int wid  = (blockIdx.x * blockDim.x + threadIdx.x) >> 5;
    const int ntn  = QKV_N / 32;                 // 96 tiles in N
    if (wid >= (S_LEN / 32) * ntn) return;
    const int rb = (wid / ntn) * 32;
    const int cb = (wid % ntn) * 32;

    floatx8 acc[2][2] = {};
    for (int kt = 0; kt < H_DIM / 32; ++kt) {
        const int k0 = kt * 32;
        bhalf16 a0 = load_a16x32(X + (size_t)rb * H_DIM + k0, H_DIM, lane);
        bhalf16 a1 = load_a16x32(X + (size_t)(rb + 16) * H_DIM + k0, H_DIM, lane);
        bhalf16 b0 = load_b32x16(W + (size_t)cb * H_DIM + k0, H_DIM, lane);
        bhalf16 b1 = load_b32x16(W + (size_t)(cb + 16) * H_DIM + k0, H_DIM, lane);
        acc[0][0] = wmma_bf16(a0, b0, acc[0][0]);
        acc[0][1] = wmma_bf16(a0, b1, acc[0][1]);
        acc[1][0] = wmma_bf16(a1, b0, acc[1][0]);
        acc[1][1] = wmma_bf16(a1, b1, acc[1][1]);
    }

    const int nl = lane & 15, half = (lane >> 4) & 1;
#pragma unroll
    for (int mi2 = 0; mi2 < 2; ++mi2) {
#pragma unroll
        for (int ni2 = 0; ni2 < 2; ++ni2) {
            const int col = cb + ni2 * 16 + nl;
            const float bv = bias[col];
#pragma unroll
            for (int r = 0; r < 8; ++r) {
                const int row = rb + mi2 * 16 + r + 8 * half; // C-layout row
                const float v = acc[mi2][ni2][r] + bv;
                qkv[(size_t)row * QKV_N + col] = (bhalf)v;
                if (col >= QSZ + KVSZ) {                      // V columns -> V^T
                    vT[(size_t)(col - (QSZ + KVSZ)) * S_LEN + row] = (bhalf)v;
                }
            }
        }
    }
}

// ---------------------------------------------------------------------------
// Kernel 3: causal GQA flash attention.  One wave per (head, 16 Q rows).
// Q tile 16x64, KV processed in blocks of 32 keys:
//   S = Q*K^T (4 wmma) -> online softmax (shfl reductions) ->
//   P -> bf16 via per-wave LDS transpose -> O += P*V (4 wmma)
// ---------------------------------------------------------------------------
__global__ void attn_kernel(const bhalf* __restrict__ qkv, const bhalf* __restrict__ vT,
                            bhalf* __restrict__ attn_out) {
    const int lane = threadIdx.x & 31;
    const int warp = threadIdx.x >> 5;                         // 0..3
    const int wid  = blockIdx.x * (blockDim.x >> 5) + warp;    // 0..4095
    const int head = wid >> 7;                                 // / (S/16)
    const int qb   = (wid & 127) * 16;
    const int g    = head >> 2;                                // NH/NG = 4
    const float scale = 0.125f;                                // 1/sqrt(64)

    __shared__ bhalf plds[4][16][32];                          // per-wave P tile

    const int nl = lane & 15, half = (lane >> 4) & 1;

    // Q as two 16x32 A fragments (d = 0..31, 32..63)
    const bhalf* qbase = qkv + (size_t)qb * QKV_N + head * HD;
    bhalf16 aq0 = load_a16x32(qbase + 0,  QKV_N, lane);
    bhalf16 aq1 = load_a16x32(qbase + 32, QKV_N, lane);

    floatx8 o[4] = {};
    float mi[8], li[8];
#pragma unroll
    for (int r = 0; r < 8; ++r) { mi[r] = -INFINITY; li[r] = 0.0f; }

    const int nblk = (qb + 16 + 31) / 32;                      // causal coverage
    for (int j = 0; j < nblk; ++j) {
        const int kvb = j * 32;
        const bhalf* kb = qkv + QSZ + (size_t)g * HD;          // K columns base
        bhalf16 bk00 = load_b32x16(kb + (size_t)(kvb +  0) * QKV_N + 0,  QKV_N, lane);
        bhalf16 bk01 = load_b32x16(kb + (size_t)(kvb +  0) * QKV_N + 32, QKV_N, lane);
        bhalf16 bk10 = load_b32x16(kb + (size_t)(kvb + 16) * QKV_N + 0,  QKV_N, lane);
        bhalf16 bk11 = load_b32x16(kb + (size_t)(kvb + 16) * QKV_N + 32, QKV_N, lane);

        floatx8 s0 = {}, s1 = {};
        s0 = wmma_bf16(aq0, bk00, s0);
        s0 = wmma_bf16(aq1, bk01, s0);
        s1 = wmma_bf16(aq0, bk10, s1);
        s1 = wmma_bf16(aq1, bk11, s1);

        // Online softmax; rows live in (vgpr r, lane-half), cols across 16 lanes.
        // xor-shuffles with mask<16 never cross the 16-lane halves.
#pragma unroll
        for (int r = 0; r < 8; ++r) {
            const int row = qb + r + 8 * half;
            const int c0 = kvb + nl, c1 = kvb + 16 + nl;
            float v0 = (c0 <= row) ? s0[r] * scale : -INFINITY;
            float v1 = (c1 <= row) ? s1[r] * scale : -INFINITY;
            float mnew = fmaxf(v0, v1);
#pragma unroll
            for (int d = 8; d >= 1; d >>= 1) mnew = fmaxf(mnew, __shfl_xor(mnew, d, 32));
            mnew = fmaxf(mi[r], mnew);
            const float corr = __expf(mi[r] - mnew);
            const float p0 = __expf(v0 - mnew);
            const float p1 = __expf(v1 - mnew);
            float rs = p0 + p1;
#pragma unroll
            for (int d = 8; d >= 1; d >>= 1) rs += __shfl_xor(rs, d, 32);
            li[r] = li[r] * corr + rs;
            mi[r] = mnew;
            o[0][r] *= corr; o[1][r] *= corr; o[2][r] *= corr; o[3][r] *= corr;
            plds[warp][r + 8 * half][nl]      = (bhalf)p0;
            plds[warp][r + 8 * half][16 + nl] = (bhalf)p1;
        }
        // per-wave LDS RAW: wait on DScnt (no cross-wave barrier needed)
        asm volatile("s_wait_dscnt 0" ::: "memory");
        bhalf16 ap = load_a16x32(&plds[warp][0][0], 32, lane);

        // V block 32x64 as four 32x16 B fragments from V^T (contiguous in s)
        const bhalf* vb = vT + (size_t)g * HD * S_LEN + kvb;
#pragma unroll
        for (int dt = 0; dt < 4; ++dt) {
            bhalf16 bv = load_b32x16(vb + (size_t)dt * 16 * S_LEN, S_LEN, lane);
            o[dt] = wmma_bf16(ap, bv, o[dt]);
        }
    }

    // Normalize and write attention output in [s, h*HD + d] layout.
#pragma unroll
    for (int r = 0; r < 8; ++r) {
        const int row = qb + r + 8 * half;
        const float inv = 1.0f / li[r];
        bhalf* orow = attn_out + (size_t)row * H_DIM + head * HD;
#pragma unroll
        for (int dt = 0; dt < 4; ++dt)
            orow[dt * 16 + nl] = (bhalf)(o[dt][r] * inv);
    }
}

// ---------------------------------------------------------------------------
// Kernel 4: out = attn @ Wo^T + bo  (fp32 output)
// ---------------------------------------------------------------------------
__global__ void gemm_out_kernel(const bhalf* __restrict__ X, const bhalf* __restrict__ W,
                                const float* __restrict__ bias, float* __restrict__ out) {
    const int lane = threadIdx.x & 31;
    const int wid  = (blockIdx.x * blockDim.x + threadIdx.x) >> 5;
    const int ntn  = H_DIM / 32;                  // 64
    if (wid >= (S_LEN / 32) * ntn) return;
    const int rb = (wid / ntn) * 32;
    const int cb = (wid % ntn) * 32;

    floatx8 acc[2][2] = {};
    for (int kt = 0; kt < H_DIM / 32; ++kt) {
        const int k0 = kt * 32;
        bhalf16 a0 = load_a16x32(X + (size_t)rb * H_DIM + k0, H_DIM, lane);
        bhalf16 a1 = load_a16x32(X + (size_t)(rb + 16) * H_DIM + k0, H_DIM, lane);
        bhalf16 b0 = load_b32x16(W + (size_t)cb * H_DIM + k0, H_DIM, lane);
        bhalf16 b1 = load_b32x16(W + (size_t)(cb + 16) * H_DIM + k0, H_DIM, lane);
        acc[0][0] = wmma_bf16(a0, b0, acc[0][0]);
        acc[0][1] = wmma_bf16(a0, b1, acc[0][1]);
        acc[1][0] = wmma_bf16(a1, b0, acc[1][0]);
        acc[1][1] = wmma_bf16(a1, b1, acc[1][1]);
    }

    const int nl = lane & 15, half = (lane >> 4) & 1;
#pragma unroll
    for (int mi2 = 0; mi2 < 2; ++mi2) {
#pragma unroll
        for (int ni2 = 0; ni2 < 2; ++ni2) {
            const int col = cb + ni2 * 16 + nl;
            const float bv = bias[col];
#pragma unroll
            for (int r = 0; r < 8; ++r) {
                const int row = rb + mi2 * 16 + r + 8 * half;
                out[(size_t)row * H_DIM + col] = acc[mi2][ni2][r] + bv;
            }
        }
    }
}

// ---------------------------------------------------------------------------
// Host launcher
// ---------------------------------------------------------------------------
extern "C" void kernel_launch(void* const* d_in, const int* in_sizes, int n_in,
                              void* d_out, int out_size, void* d_ws, size_t ws_size,
                              hipStream_t stream) {
    const float* hs   = (const float*)d_in[0];  // [1, S, H]
    const float* Wqkv = (const float*)d_in[1];  // [3072, H]
    const float* bqkv = (const float*)d_in[2];  // [3072]
    const float* Wo   = (const float*)d_in[3];  // [H, H]
    const float* bo   = (const float*)d_in[4];  // [H]
    float* out = (float*)d_out;

    char* ws = (char*)d_ws;
    bhalf* hsb   = (bhalf*)ws; ws += (size_t)S_LEN * H_DIM * sizeof(bhalf);
    bhalf* wqkvb = (bhalf*)ws; ws += (size_t)QKV_N * H_DIM * sizeof(bhalf);
    bhalf* wob   = (bhalf*)ws; ws += (size_t)H_DIM * H_DIM * sizeof(bhalf);
    bhalf* qkv   = (bhalf*)ws; ws += (size_t)S_LEN * QKV_N * sizeof(bhalf);
    bhalf* vT    = (bhalf*)ws; ws += (size_t)KVSZ  * S_LEN * sizeof(bhalf);
    bhalf* attn  = (bhalf*)ws; ws += (size_t)S_LEN * H_DIM * sizeof(bhalf);

    cast_f32_bf16<<<1024, 256, 0, stream>>>(hs,   hsb,   S_LEN * H_DIM);
    cast_f32_bf16<<<1024, 256, 0, stream>>>(Wqkv, wqkvb, QKV_N * H_DIM);
    cast_f32_bf16<<<1024, 256, 0, stream>>>(Wo,   wob,   H_DIM * H_DIM);

    // 64 x 96 = 6144 waves, 8 waves (256 threads) per block -> 768 blocks
    gemm_qkv_kernel<<<768, 256, 0, stream>>>(hsb, wqkvb, bqkv, qkv, vT);

    // 32 heads * 128 q-tiles = 4096 waves, 4 waves per block -> 1024 blocks
    attn_kernel<<<1024, 128, 0, stream>>>(qkv, vT, attn);

    // 64 x 64 = 4096 waves -> 512 blocks of 256 threads
    gemm_out_kernel<<<512, 256, 0, stream>>>(attn, wob, bo, out);
}
